// central_loss_62569083568815
// MI455X (gfx1250) — compile-verified, matrix operand order checked by reading, not run on hardware
//
#include <hip/hip_runtime.h>
#include <hip/hip_bf16.h>

typedef __attribute__((ext_vector_type(16))) _Float16 v16h;
typedef __attribute__((ext_vector_type(8)))  float    v8f;
typedef int v2i __attribute__((__vector_size__(8)));   // element type of async b64 builtins

#define BN 4
#define DN 128
#define HN 192
#define WN 192
#define WW 6                               // 192 bits / 32 = 6 words per row
#define NROWS (BN * DN * HN)               // 98304 rows per mask
#define NWORD (NROWS * WW)                 // 589824 words per mask
#define NVOX  (NROWS * WN)                 // 18874368 voxels per mask
#define KITER 8

#define AS1 __attribute__((address_space(1)))
#define AS3 __attribute__((address_space(3)))

#ifndef __has_builtin
#define __has_builtin(x) 0
#endif

#if __has_builtin(__builtin_amdgcn_global_load_async_to_lds_b64) && \
    __has_builtin(__builtin_amdgcn_global_store_async_from_lds_b64)
#define HAVE_ASYNC_LDS 1
#else
#define HAVE_ASYNC_LDS 0
#endif

#if HAVE_ASYNC_LDS
#if __has_builtin(__builtin_amdgcn_s_wait_asynccnt)
#define WAIT_ASYNC0() __builtin_amdgcn_s_wait_asynccnt(0)
#else
#define WAIT_ASYNC0() asm volatile("s_wait_asynccnt 0x0" ::: "memory")
#endif
#endif

// ---------------------------------------------------------------------------
// Stage 1: threshold + bitpack both masks (m=0: y_pred>0.4, m=1: round(y_true))
// One thread per voxel; wave32 __ballot packs 32 voxels -> 1 word. Also zeroes sk.
// ---------------------------------------------------------------------------
__global__ void __launch_bounds__(256) pack_kernel(const float* __restrict__ yp,
                                                   const float* __restrict__ yt,
                                                   unsigned* __restrict__ e,
                                                   unsigned* __restrict__ sk) {
  unsigned tid = blockIdx.x * 256u + threadIdx.x;     // < 2*NVOX
  unsigned m = tid / (unsigned)NVOX;
  unsigned v = tid - m * (unsigned)NVOX;
  // speculative prefetch of upcoming cache lines (emits global_prefetch_b8)
  __builtin_prefetch(yp + v + 65536, 0, 0);
  __builtin_prefetch(yt + v + 65536, 0, 0);
  bool bit;
  if (m == 0) bit = (yp[v] > 0.4f);
  else        bit = (rintf(yt[v]) > 0.5f);
  unsigned mask = (unsigned)__ballot(bit);            // wave32: 32-bit mask
  if ((threadIdx.x & 31u) == 0u) {
    unsigned w = tid >> 5;                            // == m*NWORD + v/32 (NVOX%32==0)
    e[w]  = mask;
    sk[w] = 0u;
  }
}

__global__ void init_kernel(unsigned* cnt) {
  if (threadIdx.x < 3) cnt[threadIdx.x] = 0u;
}

// ---------------------------------------------------------------------------
// Stage 2: one thinning iteration, fused in LDS.
//   er = erode(e); opened = dilate(er); sk |= e & ~opened; e_out = er
// Tile: interior 8(d) x 32(h) rows; e loaded with halo 2, er computed with halo 1.
// Tile fill uses gfx1250 async global->LDS copies (ASYNCcnt) when available.
// ---------------------------------------------------------------------------
__global__ void __launch_bounds__(256) morph_kernel(const unsigned* __restrict__ ein,
                                                    unsigned* __restrict__ eout,
                                                    unsigned* __restrict__ sk) {
  __shared__ unsigned ebuf[12][36][WW];   // d0-2..d0+9, h0-2..h0+33
  __shared__ unsigned erbuf[10][34][WW];  // d0-1..d0+8, h0-1..h0+32

  unsigned bid = blockIdx.x;
  int yt = (int)(bid % 6u);
  int zt = (int)((bid / 6u) % 16u);
  int b  = (int)((bid / 96u) % 4u);
  int m  = (int)(bid / 384u);
  int d0 = zt * 8, h0 = yt * 32;
  unsigned mbase = (unsigned)m * (unsigned)NWORD;

  // --- load e tile (halo 2), zero rows outside the volume (background pad) ---
  for (int r = (int)threadIdx.x; r < 12 * 36; r += 256) {
    int lz = r / 36, ly = r % 36;
    int d = d0 + lz - 2, h = h0 + ly - 2;
    if (d >= 0 && d < DN && h >= 0 && h < HN) {
      unsigned row = ((unsigned)((b * DN + d) * HN + h)) * WW;
#if HAVE_ASYNC_LDS
      const unsigned* g = &ein[mbase + row];
      unsigned* l = &ebuf[lz][ly][0];
      // 24 B per row, 8-byte aligned on both sides: 3 x b64 async copies.
      // INST_OFFSET applies to both the global and the LDS address.
      __builtin_amdgcn_global_load_async_to_lds_b64((AS1 v2i*)g, (AS3 v2i*)l, 0, 0);
      __builtin_amdgcn_global_load_async_to_lds_b64((AS1 v2i*)g, (AS3 v2i*)l, 8, 0);
      __builtin_amdgcn_global_load_async_to_lds_b64((AS1 v2i*)g, (AS3 v2i*)l, 16, 0);
#else
      #pragma unroll
      for (int j = 0; j < WW; ++j) ebuf[lz][ly][j] = ein[mbase + row + j];
#endif
    } else {
      #pragma unroll
      for (int j = 0; j < WW; ++j) ebuf[lz][ly][j] = 0u;
    }
  }
#if HAVE_ASYNC_LDS
  WAIT_ASYNC0();                       // my async LDS writes done, then barrier
#endif
  __syncthreads();

  // --- erode (27-neighborhood AND with x-shift carries) into erbuf (halo 1) ---
  for (int r = (int)threadIdx.x; r < 10 * 34; r += 256) {
    int lz = r / 34, ly = r % 34;
    unsigned acc[WW];
    #pragma unroll
    for (int j = 0; j < WW; ++j) acc[j] = 0xFFFFFFFFu;
    #pragma unroll
    for (int dz = 0; dz < 3; ++dz)
      #pragma unroll
      for (int dy = 0; dy < 3; ++dy) {
        const unsigned* row = &ebuf[lz + dz][ly + dy][0];
        #pragma unroll
        for (int j = 0; j < WW; ++j) {
          unsigned c = row[j];
          unsigned p = (j > 0)      ? row[j - 1] : 0u;
          unsigned n = (j < WW - 1) ? row[j + 1] : 0u;
          acc[j] &= c & ((c << 1) | (p >> 31)) & ((c >> 1) | (n << 31));
        }
      }
    #pragma unroll
    for (int j = 0; j < WW; ++j) erbuf[lz][ly][j] = acc[j];
  }
  __syncthreads();

  // --- interior: opened = dilate(er); sk |= e & ~opened; write er out ---
  {
    int lz = (int)(threadIdx.x / 32u), ly = (int)(threadIdx.x % 32u);
    int d = d0 + lz, h = h0 + ly;
    unsigned opened[WW];
    #pragma unroll
    for (int j = 0; j < WW; ++j) opened[j] = 0u;
    #pragma unroll
    for (int dz = 0; dz < 3; ++dz)
      #pragma unroll
      for (int dy = 0; dy < 3; ++dy) {
        const unsigned* row = &erbuf[lz + dz][ly + dy][0];
        #pragma unroll
        for (int j = 0; j < WW; ++j) {
          unsigned c = row[j];
          unsigned p = (j > 0)      ? row[j - 1] : 0u;
          unsigned n = (j < WW - 1) ? row[j + 1] : 0u;
          opened[j] |= c | ((c << 1) | (p >> 31)) | ((c >> 1) | (n << 31));
        }
      }
    unsigned gw = mbase + ((unsigned)((b * DN + d) * HN + h)) * WW;
    #pragma unroll
    for (int j = 0; j < WW; ++j) {
      unsigned e_c = ebuf[lz + 2][ly + 2][j];
      sk[gw + j] = sk[gw + j] | (e_c & ~opened[j]);
    }
#if HAVE_ASYNC_LDS
    // e_out = er: async LDS -> global store, completed by S_ENDPGM's wait-idle.
    unsigned* gout = &eout[gw];
    unsigned* l = &erbuf[lz + 1][ly + 1][0];
    __builtin_amdgcn_global_store_async_from_lds_b64((AS1 v2i*)gout, (AS3 v2i*)l, 0, 0);
    __builtin_amdgcn_global_store_async_from_lds_b64((AS1 v2i*)gout, (AS3 v2i*)l, 8, 0);
    __builtin_amdgcn_global_store_async_from_lds_b64((AS1 v2i*)gout, (AS3 v2i*)l, 16, 0);
#else
    #pragma unroll
    for (int j = 0; j < WW; ++j) eout[gw + j] = erbuf[lz + 1][ly + 1][j];
#endif
  }
}

// ---------------------------------------------------------------------------
// Stage 3: popcount sums i, j, intersection. Wave-level reduction via
// V_WMMA_F32_16X16X32_F16 (A holds one partial per lane, B = ones -> row sums;
// column sum = lane L + lane L+16 of the 8 accumulator VGPRs = exact wave total).
// ---------------------------------------------------------------------------
__global__ void __launch_bounds__(256) reduce_kernel(const unsigned* __restrict__ sk,
                                                     unsigned* __restrict__ cnt) {
  unsigned gtid = blockIdx.x * 256u + threadIdx.x;
  unsigned w0 = gtid * 2u;                      // grid covers NWORD/2 threads exactly
  unsigned a0 = sk[w0], a1 = sk[w0 + 1];
  unsigned b0 = sk[(unsigned)NWORD + w0], b1 = sk[(unsigned)NWORD + w0 + 1];
  float vals[3];
  vals[0] = (float)(__popc(a0) + __popc(a1));            // i  partial
  vals[1] = (float)(__popc(b0) + __popc(b1));            // j  partial
  vals[2] = (float)(__popc(a0 & b0) + __popc(a1 & b1));  // intersection partial

  v16h ones;
  #pragma unroll
  for (int k = 0; k < 16; ++k) ones[k] = (_Float16)1.0f;

  #pragma unroll
  for (int q = 0; q < 3; ++q) {
    v16h a;
    #pragma unroll
    for (int k = 0; k < 16; ++k) a[k] = (_Float16)0.0f;
    a[0] = (_Float16)vals[q];                 // each lane contributes one A element
    v8f c = {};
    v8f dm = __builtin_amdgcn_wmma_f32_16x16x32_f16(
        /*neg_a=*/false, a, /*neg_b=*/false, ones,
        /*c_mod=*/(short)0, c, /*reuse_a=*/false, /*reuse_b=*/false);
    float s = dm[0] + dm[1] + dm[2] + dm[3] + dm[4] + dm[5] + dm[6] + dm[7];
    float t = s + __shfl_xor(s, 16, 32);      // column sum = exact wave total
    if ((threadIdx.x & 31u) == 0u)
      atomicAdd(&cnt[q], (unsigned)(t + 0.5f));
  }
}

__global__ void final_kernel(const unsigned* __restrict__ cnt, float* __restrict__ out) {
  float i = (float)cnt[0], j = (float)cnt[1], x = (float)cnt[2];
  out[0] = 1.0f - 2.0f * x / (i + j);
}

// ---------------------------------------------------------------------------
extern "C" void kernel_launch(void* const* d_in, const int* in_sizes, int n_in,
                              void* d_out, int out_size, void* d_ws, size_t ws_size,
                              hipStream_t stream) {
  (void)in_sizes; (void)n_in; (void)out_size; (void)ws_size;
  const float* yp = (const float*)d_in[0];
  const float* yt = (const float*)d_in[1];

  unsigned char* ws = (unsigned char*)d_ws;
  const size_t S = (size_t)2 * NWORD * sizeof(unsigned);   // one double-mask buffer
  unsigned* e0  = (unsigned*)(ws);
  unsigned* e1  = (unsigned*)(ws + S);
  unsigned* sk  = (unsigned*)(ws + 2 * S);
  unsigned* cnt = (unsigned*)(ws + 3 * S);

  pack_kernel<<<(2u * NVOX) / 256u, 256, 0, stream>>>(yp, yt, e0, sk);
  init_kernel<<<1, 32, 0, stream>>>(cnt);

  unsigned* ein = e0; unsigned* eout = e1;
  for (int it = 0; it < KITER; ++it) {
    morph_kernel<<<2 * 4 * 16 * 6, 256, 0, stream>>>(ein, eout, sk);  // 768 blocks
    unsigned* t = ein; ein = eout; eout = t;
  }

  reduce_kernel<<<NWORD / 512, 256, 0, stream>>>(sk, cnt);            // 1152 blocks
  final_kernel<<<1, 1, 0, stream>>>(cnt, (float*)d_out);
}
